// EGNNEncoder_22720376996002
// MI455X (gfx1250) — matrix operand order repaired and checked by reading, not compiled, not run on hardware
//
#include <hip/hip_runtime.h>
#include <hip/hip_bf16.h>

// ---------- types ----------
typedef _Float16 v16h __attribute__((ext_vector_type(16)));
typedef _Float16 v8h  __attribute__((ext_vector_type(8)));
typedef float    v8f  __attribute__((ext_vector_type(8)));

#define NN 50000
#define NE 800000
#define TS 168          // edge-feature tile row stride (halves), 133 padded->160, +8 pad
#define EDGE_BLOCKS 1250  // grid-stride: 5000 waves, 50000 tiles -> 10 tiles/wave

__device__ inline float silu_f(float v) { return v / (1.0f + __expf(-v)); }

__device__ inline v16h ld16(const _Float16* p0, const _Float16* p1) {
    v8h lo = *(const v8h*)p0;
    v8h hi = *(const v8h*)p1;
    v16h r;
#pragma unroll
    for (int i = 0; i < 8; ++i) { r[i] = lo[i]; r[8 + i] = hi[i]; }
    return r;
}

__device__ inline v8f wmma_f16(v16h a, v16h b, v8f c) {
    return __builtin_amdgcn_wmma_f32_16x16x32_f16(false, a, false, b, (short)0, c, false, false);
}

__device__ inline void zeroC(v8f C[4]) {
#pragma unroll
    for (int nt = 0; nt < 4; ++nt)
#pragma unroll
        for (int i = 0; i < 8; ++i) C[nt][i] = 0.0f;
}

// GEMM: D(16 x 64) += A(16 x 32*KSTEPS) * B^T, B stored as Bt[n][K] (K = 32*KSTEPS)
// A layout (16-bit, 16x32/step): lane<16: M=lane, K {kb..kb+7} U {kb+16..kb+23}, kb=0
//                                lane>=16: same M set, kb=8
// B layout: same K sets as A (kb=0 for lane<16, kb=8 for lane>=16), N=lane&15
template <int KSTEPS>
__device__ inline void gemm4(const _Float16* __restrict__ A, int as,
                             const _Float16* __restrict__ Bt, int K,
                             int lane, v8f C[4]) {
    const int mA  = lane & 15;
    const int kbA = (lane >> 4) * 8;
    const int kbB = (lane >> 4) * 8;
    const int nl  = lane & 15;
#pragma unroll
    for (int ks = 0; ks < KSTEPS; ++ks) {
        const int k0 = ks * 32;
        const _Float16* pa = A + mA * as + k0 + kbA;
        v16h a = ld16(pa, pa + 16);
#pragma unroll
        for (int nt = 0; nt < 4; ++nt) {
            const _Float16* pb = Bt + (size_t)(nt * 16 + nl) * K + k0 + kbB;
            v16h b = ld16(pb, pb + 16);
            C[nt] = wmma_f16(a, b, C[nt]);
        }
    }
}

__device__ inline void stage_lds(const _Float16* __restrict__ g, _Float16* s,
                                 int halves, int tid, int nthr) {
    const uint4* gp = (const uint4*)g;
    uint4* sp = (uint4*)s;
    int n16 = halves >> 3;
    for (int i = tid; i < n16; i += nthr) sp[i] = gp[i];
}

// ---------- weight convert: Wt[n*K+k] = (k<din) ? W[k*dout+n] : 0 ----------
__global__ void convt_kernel(const float* __restrict__ w, _Float16* __restrict__ wt,
                             int din, int dout, int K) {
    int i = blockIdx.x * blockDim.x + threadIdx.x;
    if (i >= dout * K) return;
    int n = i / K, k = i % K;
    wt[i] = (k < din) ? (_Float16)w[(size_t)k * dout + n] : (_Float16)0.0f;
}

// ---------- small utility kernels ----------
__global__ void zero_kernel(float* p, int n) {
    int i = blockIdx.x * blockDim.x + threadIdx.x;
    if (i < n) p[i] = 0.0f;
}
__global__ void copy_kernel(float* d, const float* s, int n) {
    int i = blockIdx.x * blockDim.x + threadIdx.x;
    if (i < n) d[i] = s[i];
}
__global__ void count_kernel(const int* __restrict__ row, float* cnt, int n) {
    int i = blockIdx.x * blockDim.x + threadIdx.x;
    if (i < n) atomicAdd(&cnt[row[i]], 1.0f);
}
__global__ void inv_kernel(const float* cnt, float* ic, int n) {
    int i = blockIdx.x * blockDim.x + threadIdx.x;
    if (i < n) ic[i] = 1.0f / fmaxf(cnt[i], 1.0f);
}

// ---------- h = hin(11) @ W(11x64) + b ; thread computes 16 outputs ----------
__global__ void embed_kernel(const float* __restrict__ hin, const float* __restrict__ w,
                             const float* __restrict__ b, float* __restrict__ h,
                             _Float16* __restrict__ h16) {
    int t = blockIdx.x * blockDim.x + threadIdx.x;
    if (t >= NN * 4) return;
    int node = t >> 2, q = (t & 3) * 16;
    float acc[16];
#pragma unroll
    for (int j = 0; j < 16; ++j) acc[j] = b[q + j];
#pragma unroll
    for (int k = 0; k < 11; ++k) {
        float hv = hin[node * 11 + k];
        const float* wr = &w[k * 64 + q];
#pragma unroll
        for (int j = 0; j < 16; ++j) acc[j] += hv * wr[j];
    }
#pragma unroll
    for (int j = 0; j < 16; ++j) {
        h[(size_t)node * 64 + q + j] = acc[j];
        h16[(size_t)node * 64 + q + j] = (_Float16)acc[j];
    }
}

// ---------- fused per-edge MLP (grid-stride over 16-edge wave tiles):
// e1 -> silu -> e2 -> silu(+agg atomics) -> c1 -> silu -> c2 -> trans atomics
// waves_per_eu(4) + opaque weight-base offset: B fragments are re-read from LDS
// every iteration (ds_load_b128 -> v_wmma pipelining) instead of being hoisted
// into registers (which previously forced scratch spills under the VGPR cap).
__global__ __launch_bounds__(128)
__attribute__((amdgpu_waves_per_eu(4)))
void edge_kernel(
    const float* __restrict__ x, const _Float16* __restrict__ h16,
    const int* __restrict__ erow, const int* __restrict__ ecol,
    const float* __restrict__ eattr, const float* __restrict__ icnt,
    const _Float16* __restrict__ wE1, const _Float16* __restrict__ wE2,
    const _Float16* __restrict__ wC1, const _Float16* __restrict__ wC2,
    const float* __restrict__ bE1, const float* __restrict__ bE2,
    const float* __restrict__ bC1, const float* __restrict__ bC2,
    float* __restrict__ agg, float* __restrict__ tracc) {
    __shared__ __attribute__((aligned(16))) _Float16 sW1[64 * 160];
    __shared__ __attribute__((aligned(16))) _Float16 sW2[64 * 64];
    __shared__ __attribute__((aligned(16))) _Float16 sWc[64 * 64];
    __shared__ __attribute__((aligned(16))) _Float16 sWv[64];
    __shared__ __attribute__((aligned(16))) _Float16 sT[4][16 * TS];
    __shared__ int   sRow[4][16];
    __shared__ float sDiff[4][48];
    __shared__ float sIc[4][16];

    const int tid = threadIdx.x;
    stage_lds(wE1, sW1, 64 * 160, tid, 128);
    stage_lds(wE2, sW2, 64 * 64, tid, 128);
    stage_lds(wC1, sWc, 64 * 64, tid, 128);
    stage_lds(wC2, sWv, 64, tid, 128);
    __syncthreads();  // weights staged once; per-tile work below is wave-private

    const int wid = tid >> 5, lane = tid & 31;
    const int e = lane & 15, half = lane >> 4;
    const int nl = lane & 15;
    const int mbase = (lane >> 4) * 8;
    const int nwaves = gridDim.x * 4;
    _Float16* tile = &sT[wid][0];

    int zofs = 0;  // opaque 0 (refreshed per iteration) to defeat LICM of B loads
    for (int wt = blockIdx.x * 4 + wid; wt < NE / 16; wt += nwaves) {
        asm volatile("" : "+s"(zofs));
        const _Float16* w1 = sW1 + zofs;
        const _Float16* w2 = sW2 + zofs;
        const _Float16* wc = sWc + zofs;

        const int eg = wt * 16 + e;

        // prefetch next tile's index/attr streams (global_prefetch_b8)
        const int egn = eg + nwaves * 16;
        if (egn < NE) {
            __builtin_prefetch(&erow[egn], 0, 0);
            __builtin_prefetch(&ecol[egn], 0, 0);
            __builtin_prefetch(&eattr[(size_t)egn * 4], 0, 0);
        }

        // ---- gather: edge_feat tile = [h[row] | h[col] | radial | eattr | 0pad]
        {
            int r = erow[eg], c = ecol[eg];
            const uint4* hr = (const uint4*)(h16 + (size_t)r * 64 + half * 32);
            const uint4* hc = (const uint4*)(h16 + (size_t)c * 64 + half * 32);
            uint4* d0 = (uint4*)&tile[e * TS + half * 32];
            uint4* d1 = (uint4*)&tile[e * TS + 64 + half * 32];
#pragma unroll
            for (int j = 0; j < 4; ++j) d0[j] = hr[j];
#pragma unroll
            for (int j = 0; j < 4; ++j) d1[j] = hc[j];
            if (half == 0) {
                sRow[wid][e] = r;
                sIc[wid][e] = icnt[r];
                float rad = 0.0f;
#pragma unroll
                for (int d = 0; d < 3; ++d) {
                    float dd = x[(size_t)r * 3 + d] - x[(size_t)c * 3 + d];
                    sDiff[wid][e * 3 + d] = dd;
                    rad += dd * dd;
                }
                tile[e * TS + 128] = (_Float16)rad;
#pragma unroll
                for (int j = 0; j < 4; ++j)
                    tile[e * TS + 129 + j] = (_Float16)eattr[(size_t)eg * 4 + j];
                for (int j = 133; j < TS; ++j) tile[e * TS + j] = (_Float16)0.0f;
            }
        }
        // no barrier: tile is wave-private; DS ops are in-order within a wave

        v8f C[4];

        // ---- e1: (16x160)@(160x64), silu -> t1 at cols 0..63 ----
        zeroC(C);
        gemm4<5>(tile, TS, w1, 160, lane, C);
#pragma unroll
        for (int nt = 0; nt < 4; ++nt) {
            float bias = bE1[nt * 16 + nl];
#pragma unroll
            for (int r = 0; r < 8; ++r) {
                float v = silu_f(C[nt][r] + bias);
                tile[(mbase + r) * TS + nt * 16 + nl] = (_Float16)v;
            }
        }

        // ---- e2: (16x64)@(64x64), silu -> m at cols 64..127, + agg atomics ----
        zeroC(C);
        gemm4<2>(tile, TS, w2, 64, lane, C);
#pragma unroll
        for (int nt = 0; nt < 4; ++nt) {
            float bias = bE2[nt * 16 + nl];
#pragma unroll
            for (int r = 0; r < 8; ++r) {
                int m = mbase + r;
                float v = silu_f(C[nt][r] + bias);
                tile[m * TS + 64 + nt * 16 + nl] = (_Float16)v;
                atomicAdd(&agg[(size_t)sRow[wid][m] * 64 + nt * 16 + nl], v);
            }
        }

        // ---- c1: silu -> t2 at cols 0..63 ----
        zeroC(C);
        gemm4<2>(&tile[64], TS, wc, 64, lane, C);
#pragma unroll
        for (int nt = 0; nt < 4; ++nt) {
            float bias = bC1[nt * 16 + nl];
#pragma unroll
            for (int r = 0; r < 8; ++r) {
                float v = silu_f(C[nt][r] + bias);
                tile[(mbase + r) * TS + nt * 16 + nl] = (_Float16)v;
            }
        }

        // ---- c2: 64->1 dot (both lane halves, 128b LDS loads, shfl reduce) ----
        {
            float acc = 0.0f;
            const _Float16* tp = &tile[e * TS + half * 32];
            const _Float16* wp = &sWv[half * 32] + zofs;
#pragma unroll
            for (int j = 0; j < 4; ++j) {
                v8h tv = *(const v8h*)(tp + j * 8);
                v8h wv = *(const v8h*)(wp + j * 8);
#pragma unroll
                for (int i = 0; i < 8; ++i) acc += (float)tv[i] * (float)wv[i];
            }
            acc += __shfl_xor(acc, 16, 32);
            if (half == 0) {
                float s = acc + bC2[0];
                float ic = sIc[wid][e];
                int rw = sRow[wid][e];
#pragma unroll
                for (int d = 0; d < 3; ++d)
                    atomicAdd(&tracc[(size_t)rw * 3 + d], sDiff[wid][e * 3 + d] * s * ic);
            }
        }
    }
}

// ---------- node update: h += n2(silu(n1([h, agg]))), x += tracc ----------
__global__ __launch_bounds__(128) void node_kernel(
    float* __restrict__ h, _Float16* __restrict__ h16, float* __restrict__ x,
    const float* __restrict__ agg, const float* __restrict__ tracc,
    const _Float16* __restrict__ wN1, const _Float16* __restrict__ wN2,
    const float* __restrict__ bN1, const float* __restrict__ bN2) {
    __shared__ __attribute__((aligned(16))) _Float16 sW1[64 * 128];
    __shared__ __attribute__((aligned(16))) _Float16 sW2[64 * 64];
    __shared__ __attribute__((aligned(16))) _Float16 sA[4][16 * 136];
    __shared__ __attribute__((aligned(16))) _Float16 sM[4][16 * 72];

    const int tid = threadIdx.x;
    stage_lds(wN1, sW1, 64 * 128, tid, 128);
    stage_lds(wN2, sW2, 64 * 64, tid, 128);

    const int wid = tid >> 5, lane = tid & 31;
    const int e = lane & 15, half = lane >> 4;
    const int gw = blockIdx.x * 4 + wid;
    const bool act = (gw < (NN / 16));
    const int nb = gw * 16;

    if (act) {
        int node = nb + e;
        const uint4* hp = (const uint4*)(h16 + (size_t)node * 64 + half * 32);
        uint4* dd = (uint4*)&sA[wid][e * 136 + half * 32];
#pragma unroll
        for (int j = 0; j < 4; ++j) dd[j] = hp[j];
#pragma unroll
        for (int j = 0; j < 8; ++j) {
            float4 av = *(const float4*)&agg[(size_t)node * 64 + half * 32 + j * 4];
            _Float16* dp = &sA[wid][e * 136 + 64 + half * 32 + j * 4];
            dp[0] = (_Float16)av.x; dp[1] = (_Float16)av.y;
            dp[2] = (_Float16)av.z; dp[3] = (_Float16)av.w;
        }
    }
    __syncthreads();
    if (!act) return;

    const int nl = lane & 15;
    const int mbase = (lane >> 4) * 8;
    v8f C[4];

    // n1: (16x128)@(128x64), silu -> sM
    zeroC(C);
    gemm4<4>(&sA[wid][0], 136, sW1, 128, lane, C);
#pragma unroll
    for (int nt = 0; nt < 4; ++nt) {
        float bias = bN1[nt * 16 + nl];
#pragma unroll
        for (int r = 0; r < 8; ++r) {
            float v = silu_f(C[nt][r] + bias);
            sM[wid][(mbase + r) * 72 + nt * 16 + nl] = (_Float16)v;
        }
    }

    // n2: (16x64)@(64x64), residual h update
    zeroC(C);
    gemm4<2>(&sM[wid][0], 72, sW2, 64, lane, C);
#pragma unroll
    for (int nt = 0; nt < 4; ++nt) {
        float bias = bN2[nt * 16 + nl];
#pragma unroll
        for (int r = 0; r < 8; ++r) {
            int node = nb + mbase + r;
            size_t ix = (size_t)node * 64 + nt * 16 + nl;
            float hn = h[ix] + C[nt][r] + bias;
            h[ix] = hn;
            h16[ix] = (_Float16)hn;
        }
    }

    if (half == 0) {
        int node = nb + e;
#pragma unroll
        for (int d = 0; d < 3; ++d)
            x[(size_t)node * 3 + d] += tracc[(size_t)node * 3 + d];
    }
}

// ---------- final: h_emb = h @ Wout + b ; also emit x ----------
__global__ __launch_bounds__(128) void final_kernel(
    const _Float16* __restrict__ h16, const float* __restrict__ x,
    const _Float16* __restrict__ wO, const float* __restrict__ bO,
    float* __restrict__ out) {
    __shared__ __attribute__((aligned(16))) _Float16 sW[64 * 64];
    __shared__ __attribute__((aligned(16))) _Float16 sA[4][16 * 72];

    const int tid = threadIdx.x;
    stage_lds(wO, sW, 64 * 64, tid, 128);

    const int wid = tid >> 5, lane = tid & 31;
    const int e = lane & 15, half = lane >> 4;
    const int gw = blockIdx.x * 4 + wid;
    const bool act = (gw < (NN / 16));
    const int nb = gw * 16;

    if (act) {
        int node = nb + e;
        const uint4* hp = (const uint4*)(h16 + (size_t)node * 64 + half * 32);
        uint4* dd = (uint4*)&sA[wid][e * 72 + half * 32];
#pragma unroll
        for (int j = 0; j < 4; ++j) dd[j] = hp[j];
    }
    __syncthreads();
    if (!act) return;

    const int nl = lane & 15;
    const int mbase = (lane >> 4) * 8;
    v8f C[4];
    zeroC(C);
    gemm4<2>(&sA[wid][0], 72, sW, 64, lane, C);
#pragma unroll
    for (int nt = 0; nt < 4; ++nt) {
        float bias = bO[nt * 16 + nl];
#pragma unroll
        for (int r = 0; r < 8; ++r) {
            int node = nb + mbase + r;
            out[(size_t)node * 64 + nt * 16 + nl] = C[nt][r] + bias;
        }
    }
    if (half == 0) {
        int node = nb + e;
#pragma unroll
        for (int d = 0; d < 3; ++d)
            out[(size_t)NN * 64 + (size_t)node * 3 + d] = x[(size_t)node * 3 + d];
    }
}

// ---------- workspace layout (bytes) ----------
static const size_t OFF_H    = 0;          // 50000*64*4  = 12,800,000
static const size_t OFF_H16  = 12800000;   // 50000*64*2  =  6,400,000
static const size_t OFF_X    = 19200000;   // 50000*3*4   =    600,000
static const size_t OFF_AGG  = 19800000;   // 50000*64*4  = 12,800,000
static const size_t OFF_TR   = 32600000;   // 50000*3*4   =    600,000
static const size_t OFF_CNT  = 33200000;   // 50000*4
static const size_t OFF_ICNT = 33400000;   // 50000*4
static const size_t OFF_W16  = 33600000;   // fp16 weights

// per-layer fp16 weight offsets (in halves)
static const size_t LSTRIDE = 30784;
static const size_t WO_E1 = 0;      // 64*160
static const size_t WO_E2 = 10240;  // 64*64
static const size_t WO_C1 = 14336;  // 64*64
static const size_t WO_C2 = 18432;  // 64
static const size_t WO_N1 = 18496;  // 64*128
static const size_t WO_N2 = 26688;  // 64*64
static const size_t WO_EMBOUT = 4 * LSTRIDE;

extern "C" void kernel_launch(void* const* d_in, const int* in_sizes, int n_in,
                              void* d_out, int out_size, void* d_ws, size_t ws_size,
                              hipStream_t stream) {
    (void)in_sizes; (void)n_in; (void)out_size; (void)ws_size;

    const float* hin   = (const float*)d_in[0];
    const float* xin   = (const float*)d_in[1];
    const int*   ei    = (const int*)d_in[2];     // row = ei[0..E), col = ei[E..2E)
    const float* eattr = (const float*)d_in[3];
    // params flattened (jax pytree: sorted dict keys): emb_in{b,w}, emb_out{b,w},
    // layers[0..3]{c1{b,w}, c2{b,w}, e1{b,w}, e2{b,w}, n1{b,w}, n2{b,w}}
    auto P = [&](int i) { return (const float*)d_in[i]; };
    const int EMB_IN_B = 4, EMB_IN_W = 5, EMB_OUT_B = 6, EMB_OUT_W = 7;

    char* ws = (char*)d_ws;
    float*    H    = (float*)(ws + OFF_H);
    _Float16* H16  = (_Float16*)(ws + OFF_H16);
    float*    X    = (float*)(ws + OFF_X);
    float*    AGG  = (float*)(ws + OFF_AGG);
    float*    TR   = (float*)(ws + OFF_TR);
    float*    CNT  = (float*)(ws + OFF_CNT);
    float*    ICNT = (float*)(ws + OFF_ICNT);
    _Float16* W16  = (_Float16*)(ws + OFF_W16);

    // ---- convert / transpose weights to fp16 ----
    auto conv = [&](const float* w, _Float16* wt, int din, int dout, int K) {
        int n = dout * K;
        convt_kernel<<<(n + 255) / 256, 256, 0, stream>>>(w, wt, din, dout, K);
    };
    for (int l = 0; l < 4; ++l) {
        int L = 8 + 12 * l;
        _Float16* wl = W16 + (size_t)l * LSTRIDE;
        conv(P(L + 5), wl + WO_E1, 133, 64, 160);  // e1.w
        conv(P(L + 7), wl + WO_E2, 64, 64, 64);    // e2.w
        conv(P(L + 1), wl + WO_C1, 64, 64, 64);    // c1.w
        conv(P(L + 3), wl + WO_C2, 64, 1, 64);     // c2.w
        conv(P(L + 9), wl + WO_N1, 128, 64, 128);  // n1.w
        conv(P(L + 11), wl + WO_N2, 64, 64, 64);   // n2.w
    }
    conv(P(EMB_OUT_W), W16 + WO_EMBOUT, 64, 64, 64);

    // ---- init ----
    copy_kernel<<<(NN * 3 + 255) / 256, 256, 0, stream>>>(X, xin, NN * 3);
    embed_kernel<<<(NN * 4 + 255) / 256, 256, 0, stream>>>(hin, P(EMB_IN_W), P(EMB_IN_B), H, H16);
    zero_kernel<<<(NN + 255) / 256, 256, 0, stream>>>(CNT, NN);
    count_kernel<<<(NE + 255) / 256, 256, 0, stream>>>(ei, CNT, NE);
    inv_kernel<<<(NN + 255) / 256, 256, 0, stream>>>(CNT, ICNT, NN);

    const int nodeBlocks = (NN / 16 + 3) / 4;  // 3125 waves -> 782 blocks
    for (int l = 0; l < 4; ++l) {
        int L = 8 + 12 * l;
        _Float16* wl = W16 + (size_t)l * LSTRIDE;
        zero_kernel<<<(NN * 64 + 255) / 256, 256, 0, stream>>>(AGG, NN * 64);
        zero_kernel<<<(NN * 3 + 255) / 256, 256, 0, stream>>>(TR, NN * 3);
        edge_kernel<<<EDGE_BLOCKS, 128, 0, stream>>>(
            X, H16, ei, ei + NE, eattr, ICNT,
            wl + WO_E1, wl + WO_E2, wl + WO_C1, wl + WO_C2,
            P(L + 4), P(L + 6), P(L), P(L + 2),   // bE1, bE2, bC1, bC2
            AGG, TR);
        node_kernel<<<nodeBlocks, 128, 0, stream>>>(
            H, H16, X, AGG, TR,
            wl + WO_N1, wl + WO_N2, P(L + 8), P(L + 10));
    }
    final_kernel<<<nodeBlocks, 128, 0, stream>>>(H16, X, W16 + WO_EMBOUT,
                                                 P(EMB_OUT_B), (float*)d_out);
}
